// RNN_70420283785738
// MI455X (gfx1250) — matrix operand order repaired and checked by reading, not compiled, not run on hardware
//
#include <hip/hip_runtime.h>
#include <math.h>

// ---------------------------------------------------------------------------
// 2-layer LSTM + classifier head for MI455X (gfx1250, wave32, WMMA).
// GEMMs use v_wmma_f32_16x16x32_bf16 (bf16 inputs, fp32 accumulate).
// Per time step each block owns a 32(M) x 128(H) tile of all four gates:
//  - the 32 x Ktot A-strip ([x_t | h_prev]) is staged into LDS once per block
//    via GLOBAL_LOAD_ASYNC_TO_LDS_B128 (ASYNCcnt), then read as ds_load_b128;
//  - weights stream from global (L2-resident, ~7 MB total) and each B
//    fragment is reused for two M-strips;
//  - LSTM pointwise (sigmoid/tanh, c update) fused in registers; only h(bf16)
//    and c(fp32) are written. Gate tensors never touch memory.
// ---------------------------------------------------------------------------

typedef __attribute__((ext_vector_type(8)))  __bf16 v8bf;
typedef __attribute__((ext_vector_type(16))) __bf16 v16bf;
typedef __attribute__((ext_vector_type(8)))  float  v8f;
typedef __attribute__((__vector_size__(4 * sizeof(int)))) int v4i;

#define DEVINL __device__ __forceinline__

#if defined(__gfx1250__) && __has_builtin(__builtin_amdgcn_global_load_async_to_lds_b128)
#define HAVE_ASYNC_LDS 1
#else
#define HAVE_ASYNC_LDS 0
#endif

#define AS_GLOBAL __attribute__((address_space(1)))
#define AS_LDS    __attribute__((address_space(3)))

constexpr int BB     = 16384;   // batch
constexpr int TT     = 10;      // time steps
constexpr int HH     = 512;     // hidden
constexpr int G4H    = 2048;    // 4*H
constexpr int DPAD   = 32;      // D_IN (30) padded to 32 for K=32 WMMA
constexpr int NLBL   = 1000;    // labels
constexpr int NLBLP  = 1024;    // padded labels
constexpr int K0CAT  = DPAD + HH;   // 544  (layer-0 concat K)
constexpr int K1CAT  = HH + HH;     // 1024 (layer-1 concat K)
constexpr int KSMAX  = K1CAT + 8;   // padded LDS row stride (break 2KB stride)

// ---- WMMA fragment loaders (layouts per CDNA5 ISA 7.12.2) ------------------
// A fragment, 16x32 bf16, row-major source [row, K], stride in elements.
// lane m = lane&15 ; K half = lane>>4 : chunks {kh*8..+7} and {16+kh*8..+7}.
DEVINL v16bf load_a_frag(const __bf16* __restrict__ base, int stride,
                         int row0, int kbase, int lane) {
    const int m  = lane & 15;
    const int kh = lane >> 4;
    const __bf16* p = base + (size_t)(row0 + m) * (size_t)stride + kbase + kh * 8;
    v8bf lo = *(const v8bf*)(p);        // K = kbase + kh*8 .. +7
    v8bf hi = *(const v8bf*)(p + 16);   // K = kbase + 16 + kh*8 .. +7
    return __builtin_shufflevector(lo, hi, 0,1,2,3,4,5,6,7,8,9,10,11,12,13,14,15);
}

// B fragment, 32x16 bf16. Weight stored row-major [N, K] (PyTorch layout), so
// A @ frag == A @ W^T. Lane n = lane&15 holds column n; K half = lane>>4:
// chunks {kh*8..+7} and {16+kh*8..+7} (same as A).
DEVINL v16bf load_b_frag(const __bf16* __restrict__ base, int stride,
                         int n0, int kbase, int lane) {
    const int n  = lane & 15;
    const int kh = lane >> 4;
    const __bf16* p = base + (size_t)(n0 + n) * (size_t)stride + kbase + kh * 8;
    v8bf lo = *(const v8bf*)(p);
    v8bf hi = *(const v8bf*)(p + 16);
    return __builtin_shufflevector(lo, hi, 0,1,2,3,4,5,6,7,8,9,10,11,12,13,14,15);
}

DEVINL float sigmoidf(float x) { return 1.0f / (1.0f + __expf(-x)); }

// ---- weight / bias preparation (fp32 -> bf16, concat layouts) --------------
__global__ void prep_w0_kernel(const float* __restrict__ W_ih0,
                               const float* __restrict__ W_hh0,
                               __bf16* __restrict__ W0) {
    int idx = blockIdx.x * blockDim.x + threadIdx.x;   // [2048, 544]
    if (idx >= G4H * K0CAT) return;
    int r = idx / K0CAT, c = idx % K0CAT;
    float v = (c < 30)  ? W_ih0[r * 30 + c]
            : (c < DPAD) ? 0.0f
                         : W_hh0[r * HH + (c - DPAD)];
    W0[idx] = (__bf16)v;
}

__global__ void prep_w1_kernel(const float* __restrict__ W_ih1,
                               const float* __restrict__ W_hh1,
                               __bf16* __restrict__ W1) {
    int idx = blockIdx.x * blockDim.x + threadIdx.x;   // [2048, 1024]
    if (idx >= G4H * K1CAT) return;
    int r = idx / K1CAT, c = idx % K1CAT;
    float v = (c < HH) ? W_ih1[r * HH + c] : W_hh1[r * HH + (c - HH)];
    W1[idx] = (__bf16)v;
}

__global__ void prep_wout_kernel(const float* __restrict__ W_out,
                                 __bf16* __restrict__ Wo) {
    int idx = blockIdx.x * blockDim.x + threadIdx.x;   // [1024, 512]
    if (idx >= NLBLP * HH) return;
    int r = idx / HH, c = idx % HH;
    Wo[idx] = (__bf16)((r < NLBL) ? W_out[r * HH + c] : 0.0f);
}

__global__ void prep_bias_kernel(const float* __restrict__ b_ih0,
                                 const float* __restrict__ b_hh0,
                                 const float* __restrict__ b_ih1,
                                 const float* __restrict__ b_hh1,
                                 float* __restrict__ bias0,
                                 float* __restrict__ bias1) {
    int i = blockIdx.x * blockDim.x + threadIdx.x;
    if (i >= G4H) return;
    bias0[i] = b_ih0[i] + b_hh0[i];
    bias1[i] = b_ih1[i] + b_hh1[i];
}

// ---- embedding gather -> bf16 feats [T, B, 32] -----------------------------
__global__ void gather_kernel(const long long* __restrict__ x,   // [B,T,3] int64
                              const float* __restrict__ car,     // [20000,10]
                              const float* __restrict__ region,  // [1067,10]
                              const float* __restrict__ poi,     // [5000,10]
                              __bf16* __restrict__ feats) {      // [T,B,32]
    int i = blockIdx.x * blockDim.x + threadIdx.x;   // item id in [0, B*T)
    if (i >= BB * TT) return;
    int b = i / TT, t = i % TT;
    long long i0 = x[((size_t)b * TT + t) * 3 + 0];
    long long i1 = x[((size_t)b * TT + t) * 3 + 1];
    long long i2 = x[((size_t)b * TT + t) * 3 + 2];
    __bf16* dst = feats + ((size_t)t * BB + b) * DPAD;
    #pragma unroll
    for (int e = 0; e < 10; ++e) dst[e]      = (__bf16)car[i0 * 10 + e];
    #pragma unroll
    for (int e = 0; e < 10; ++e) dst[10 + e] = (__bf16)region[i1 * 10 + e];
    #pragma unroll
    for (int e = 0; e < 10; ++e) dst[20 + e] = (__bf16)poi[i2 * 10 + e];
    dst[30] = (__bf16)0.0f;
    dst[31] = (__bf16)0.0f;
}

// ---- fused LSTM step -------------------------------------------------------
// Grid: 2048 blocks x 256 threads (8 waves).
// Block tile: M-strip of 32 batch rows x 128 hidden cols (wave j0 = 16 cols,
// all four gates). A-strip (32 x Ktot bf16, padded stride) staged in LDS via
// async global->LDS; B fragments stream from global and feed two WMMAs each.
__global__ void __launch_bounds__(256)
lstm_step_kernel(const __bf16* __restrict__ A1, int lda1, int K1,
                 const __bf16* __restrict__ A2, int lda2, int K2,
                 const __bf16* __restrict__ W,  int ldw,
                 const float*  __restrict__ bias,
                 float*  __restrict__ c,        // [B,H] fp32 cell state (r/w)
                 __bf16* __restrict__ h_out,    // [B,H] bf16 new hidden
                 int has_prev) {
    __shared__ __align__(16) __bf16 smem[32 * KSMAX];   // 66 KB (<= 320 KB WGP)

    const int tid  = threadIdx.x;
    const int lane = tid & 31;
    const int wave = tid >> 5;
    const int m0   = (blockIdx.x >> 2) * 32;
    const int j0   = (((blockIdx.x & 3) << 3) + wave) * 16;

    const int Ktot = K1 + (has_prev ? K2 : 0);
    const int KS   = Ktot + 8;          // padded LDS row stride (elements)
    const int cpr  = Ktot >> 3;         // 16-byte chunks per row
    const int total = 32 * cpr;

    // ---- stage A-strip [32][Ktot] into LDS ----
    for (int ch = tid; ch < total; ch += 256) {
        const int row = ch / cpr;
        const int ke  = (ch - row * cpr) << 3;
        const __bf16* src = (ke < K1)
            ? A1 + (size_t)(m0 + row) * lda1 + ke
            : A2 + (size_t)(m0 + row) * lda2 + (ke - K1);
        __bf16* dst = &smem[row * KS + ke];
#if HAVE_ASYNC_LDS
        __builtin_amdgcn_global_load_async_to_lds_b128(
            (AS_GLOBAL v4i*)src, (AS_LDS v4i*)dst, 0, 0);
#else
        *(v8bf*)dst = *(const v8bf*)src;
#endif
    }
#if HAVE_ASYNC_LDS
#if __has_builtin(__builtin_amdgcn_s_wait_asynccnt)
    __builtin_amdgcn_s_wait_asynccnt(0);
#else
    asm volatile("s_wait_asynccnt 0" ::: "memory");
#endif
#endif
    __syncthreads();

    // ---- K-loop: 8 WMMAs per k-step (2 M-strips x 4 gates) ----
    v8f acc[2][4] = {};
    for (int kb = 0; kb < Ktot; kb += 32) {
        v16bf af0 = load_a_frag(smem, KS, 0,  kb, lane);   // ds_load_b128 x2
        v16bf af1 = load_a_frag(smem, KS, 16, kb, lane);
        #pragma unroll
        for (int g = 0; g < 4; ++g) {
            v16bf bf = load_b_frag(W, ldw, g * HH + j0, kb, lane);
            acc[0][g] = __builtin_amdgcn_wmma_f32_16x16x32_bf16(
                            false, af0, false, bf, (short)0, acc[0][g], false, false);
            acc[1][g] = __builtin_amdgcn_wmma_f32_16x16x32_bf16(
                            false, af1, false, bf, (short)0, acc[1][g], false, false);
        }
    }

    // ---- fused LSTM pointwise; C/D layout: VGPR r -> rows r (lanes 0-15)
    // and r+8 (lanes 16-31), column j0 + (lane&15).
    const int col = j0 + (lane & 15);
    const float bi  = bias[col];
    const float bfb = bias[col + HH];
    const float bg  = bias[col + 2 * HH];
    const float bo  = bias[col + 3 * HH];
    #pragma unroll
    for (int s = 0; s < 2; ++s) {
        const int rowbase = m0 + s * 16 + ((lane >> 4) << 3);
        #pragma unroll
        for (int r = 0; r < 8; ++r) {
            const size_t idx = (size_t)(rowbase + r) * HH + col;
            float ig = sigmoidf(acc[s][0][r] + bi);
            float fg = sigmoidf(acc[s][1][r] + bfb);
            float gg = tanhf(acc[s][2][r] + bg);
            float og = sigmoidf(acc[s][3][r] + bo);
            float cprev = has_prev ? c[idx] : 0.0f;
            float cn = fg * cprev + ig * gg;
            c[idx] = cn;
            h_out[idx] = (__bf16)(og * tanhf(cn));
        }
    }
}

// ---- classifier head: out[B,1000] = h2 @ W_out^T + b_out -------------------
// Grid: 8192 blocks x 256. Each wave computes one 16x16 tile, K=512.
__global__ void __launch_bounds__(256)
out_gemm_kernel(const __bf16* __restrict__ h2,   // [B,512] bf16
                const __bf16* __restrict__ Wo,   // [1024,512] bf16
                const float*  __restrict__ b_out,
                float* __restrict__ out) {       // [B,1000] fp32
    const int lane = threadIdx.x & 31;
    const int wave = threadIdx.x >> 5;
    const int m0 = (blockIdx.x >> 3) * 16;
    const int n0 = (((blockIdx.x & 7) << 3) + wave) * 16;

    v8f acc = {};
    for (int kb = 0; kb < HH; kb += 32) {
        v16bf af = load_a_frag(h2, HH, m0, kb, lane);
        v16bf bf = load_b_frag(Wo, HH, n0, kb, lane);
        acc = __builtin_amdgcn_wmma_f32_16x16x32_bf16(
                  false, af, false, bf, (short)0, acc, false, false);
    }
    const int col = n0 + (lane & 15);
    if (col < NLBL) {
        const float bo = b_out[col];
        const int rowbase = m0 + ((lane >> 4) << 3);
        #pragma unroll
        for (int r = 0; r < 8; ++r)
            out[(size_t)(rowbase + r) * NLBL + col] = acc[r] + bo;
    }
}

// ---------------------------------------------------------------------------
extern "C" void kernel_launch(void* const* d_in, const int* in_sizes, int n_in,
                              void* d_out, int out_size, void* d_ws, size_t ws_size,
                              hipStream_t stream) {
    (void)in_sizes; (void)n_in; (void)out_size; (void)ws_size;

    const long long* x    = (const long long*)d_in[0];   // int64 indices
    const float* car_emb  = (const float*)d_in[1];
    const float* reg_emb  = (const float*)d_in[2];
    const float* poi_emb  = (const float*)d_in[3];
    const float* W_ih0    = (const float*)d_in[4];
    const float* W_hh0    = (const float*)d_in[5];
    const float* b_ih0    = (const float*)d_in[6];
    const float* b_hh0    = (const float*)d_in[7];
    const float* W_ih1    = (const float*)d_in[8];
    const float* W_hh1    = (const float*)d_in[9];
    const float* b_ih1    = (const float*)d_in[10];
    const float* b_hh1    = (const float*)d_in[11];
    const float* W_out    = (const float*)d_in[12];
    const float* b_out    = (const float*)d_in[13];
    float* out            = (float*)d_out;

    // workspace bump allocator (256B aligned); total ~287 MB
    char*  ws  = (char*)d_ws;
    size_t off = 0;
    auto alloc = [&](size_t bytes) -> void* {
        off = (off + 255) & ~(size_t)255;
        void* p = ws + off;
        off += bytes;
        return p;
    };
    __bf16* feats  = (__bf16*)alloc((size_t)TT * BB * DPAD * 2);   // [T,B,32]
    __bf16* h1_all = (__bf16*)alloc((size_t)TT * BB * HH * 2);     // [T,B,512]
    __bf16* h2a    = (__bf16*)alloc((size_t)BB * HH * 2);
    __bf16* h2b    = (__bf16*)alloc((size_t)BB * HH * 2);
    float*  c0     = (float*) alloc((size_t)BB * HH * 4);
    float*  c1     = (float*) alloc((size_t)BB * HH * 4);
    __bf16* W0cat  = (__bf16*)alloc((size_t)G4H * K0CAT * 2);
    __bf16* W1cat  = (__bf16*)alloc((size_t)G4H * K1CAT * 2);
    __bf16* Woutb  = (__bf16*)alloc((size_t)NLBLP * HH * 2);
    float*  bias0  = (float*) alloc((size_t)G4H * 4);
    float*  bias1  = (float*) alloc((size_t)G4H * 4);

    // weight / bias prep
    prep_w0_kernel  <<<(G4H * K0CAT + 255) / 256, 256, 0, stream>>>(W_ih0, W_hh0, W0cat);
    prep_w1_kernel  <<<(G4H * K1CAT + 255) / 256, 256, 0, stream>>>(W_ih1, W_hh1, W1cat);
    prep_wout_kernel<<<(NLBLP * HH  + 255) / 256, 256, 0, stream>>>(W_out, Woutb);
    prep_bias_kernel<<<(G4H + 255) / 256, 256, 0, stream>>>(b_ih0, b_hh0, b_ih1, b_hh1,
                                                            bias0, bias1);
    // embedding gather
    gather_kernel<<<(BB * TT + 255) / 256, 256, 0, stream>>>(x, car_emb, reg_emb,
                                                             poi_emb, feats);

    const dim3 stepGrid(2048), stepBlk(256);

    // layer 0: A = [feats_t (K=32) | h1_{t-1} (K=512)], h state lives in h1_all
    for (int t = 0; t < TT; ++t) {
        const __bf16* a1 = feats + (size_t)t * BB * DPAD;
        const __bf16* a2 = (t == 0) ? nullptr : h1_all + (size_t)(t - 1) * BB * HH;
        __bf16* hO = h1_all + (size_t)t * BB * HH;
        lstm_step_kernel<<<stepGrid, stepBlk, 0, stream>>>(
            a1, DPAD, DPAD, a2, HH, HH, W0cat, K0CAT, bias0, c0, hO, t == 0 ? 0 : 1);
    }

    // layer 1: A = [h1_t (K=512) | h2_{t-1} (K=512)], ping-pong h2 buffers
    for (int t = 0; t < TT; ++t) {
        const __bf16* a1 = h1_all + (size_t)t * BB * HH;
        const __bf16* a2 = (t == 0) ? nullptr : ((t & 1) ? h2a : h2b);
        __bf16* hO = (t & 1) ? h2b : h2a;
        lstm_step_kernel<<<stepGrid, stepBlk, 0, stream>>>(
            a1, HH, HH, a2, HH, HH, W1cat, K1CAT, bias1, c1, hO, t == 0 ? 0 : 1);
    }
    const __bf16* h2_last = ((TT - 1) & 1) ? h2b : h2a;   // T=10 -> h2b

    // classifier head
    out_gemm_kernel<<<8192, 256, 0, stream>>>(h2_last, Woutb, b_out, out);
}
